// GlobalRetrievalBranch_42056319762525
// MI455X (gfx1250) — matrix operand order repaired and checked by reading, not compiled, not run on hardware
//
#include <hip/hip_runtime.h>
#include <hip/hip_bf16.h>

// ---------------------------------------------------------------------------
// Problem constants (from reference): N=32 batches, C=256 channels,
// HW=64 -> 4096 pixels/batch, K=128 clusters, S=8 -> 64 blocks of 64 pixels,
// output (32, 64, 129) fp32.
// ---------------------------------------------------------------------------
#define NBATCH 32
#define NCHAN  256
#define NPIX   4096   // 64*64
#define NCLUS  128
#define OUTEL  (32 * 64 * 129)

typedef __attribute__((ext_vector_type(16))) __bf16 v16bf;
typedef __attribute__((ext_vector_type(8)))  float  v8f;

// ws layout: [0,512)            c2[k]          (128 floats)
//            [512, 512+64K)     Bpack_hi       (32768 bf16)
//            [512+64K, +128K)   Bpack_lo       (32768 bf16)
#define WS_BH_OFF 512
#define WS_BL_OFF (512 + 65536)

// ---------------------------------------------------------------------------
// c2[k] = sum_c centers[k][c]^2   (exact fp32)
// ---------------------------------------------------------------------------
__global__ void grb_prep_c2(const float* __restrict__ cc, float* __restrict__ c2) {
    int k = threadIdx.x;                    // 128 threads
    const float* p = cc + k * NCHAN;
    float s = 0.0f;
    for (int c = 0; c < NCHAN; ++c) { float v = p[c]; s += v * v; }
    c2[k] = s;
}

// ---------------------------------------------------------------------------
// Pack centers into per-lane WMMA B-fragment order (bf16 hi/lo split).
// B is 32x16 (K=channels x N=clusters) per (chunk, ntile).
// B k order used here (contiguous, matching A; not the ISA table order):
// lane group 0 (lanes 0-15) holds K=0..15 packed 2/VGPR, group 1
// holds K=16..31; lane%16 = N column.
// Element index e = (((chunk*8 + ntile)*32 + lane)*16 + j),
//   channel = chunk*32 + (lane/16)*16 + j,  cluster = ntile*16 + lane%16.
// ---------------------------------------------------------------------------
__global__ void grb_prep_bpack(const float* __restrict__ cc,
                               __bf16* __restrict__ bh, __bf16* __restrict__ bl) {
    int e = blockIdx.x * 256 + threadIdx.x;          // 32768 total
    if (e >= 8 * 8 * 32 * 16) return;
    int j    = e & 15;
    int lane = (e >> 4) & 31;
    int t    = (e >> 9) & 7;
    int ch   = (e >> 12);
    int channel = ch * 32 + (lane >> 4) * 16 + j;
    int cluster = t * 16 + (lane & 15);
    float v  = cc[cluster * NCHAN + channel];
    __bf16 h = (__bf16)v;                            // RNE truncation
    bh[e] = h;
    bl[e] = (__bf16)(v - (float)h);                  // residual
}

// ---------------------------------------------------------------------------
// Zero the output (harness poisons it with 0xAA; we accumulate atomically).
// ---------------------------------------------------------------------------
__global__ void grb_zero_out(float* __restrict__ out, int n) {
    int i = blockIdx.x * 256 + threadIdx.x;
    if (i < n) out[i] = 0.0f;
}

// ---------------------------------------------------------------------------
// Main kernel: one wave = 16 consecutive pixels x all 128 clusters.
//   acc[t] (t=0..7) : 16x16 f32 WMMA tiles, D[pixel][cluster].
//   Per channel chunk of 32: build A fragments (bf16 hi/lo split of x),
//   then 8 ntiles x 3 WMMAs (hi*hi + hi*lo + lo*hi).
//   Argmin of c2[k] - 2*acc over k, then atomic histogram add.
// ---------------------------------------------------------------------------
__global__ __launch_bounds__(256) void grb_vq_main(
    const float* __restrict__ x,
    const float* __restrict__ c2,
    const __bf16* __restrict__ bh,
    const __bf16* __restrict__ bl,
    float* __restrict__ out) {

    const int lane = threadIdx.x & 31;
    const int wave = threadIdx.x >> 5;
    const int tile = blockIdx.x * 8 + wave;          // 0..8191
    const int n    = tile >> 8;                      // batch
    const int tp   = tile & 255;                     // pixel tile in batch
    const int p0   = tp << 4;                        // first pixel
    const int h    = tp >> 2;                        // row (HW=64)
    const int w0   = (tp & 3) << 4;                  // first col

    const int g   = lane >> 4;                       // lane half-group
    const int col = lane & 15;                       // N column / pixel-in-group

    v8f acc[8] = {};                                 // 8 cluster tiles

    // Base pointer for this lane's pixel within batch n.
    const float* xp = x + ((size_t)n * NCHAN) * NPIX + (size_t)(p0 + col);

    #pragma unroll
    for (int cc = 0; cc < 8; ++cc) {                 // channel chunks of 32
        const float* xc0 = xp + (size_t)(cc * 32) * NPIX;

        // Prefetch next chunk (speculative; OOB silently dropped).
        __builtin_prefetch(xc0 + 32 * NPIX, 0, 1);

        // Build A fragments: contiguous k order (matches Bpack, not ISA table):
        // group0 lanes: K = {0..15}; group1: K = {16..31}.
        v16bf a_hi, a_lo;
        #pragma unroll
        for (int j = 0; j < 16; ++j) {
            const int kk = g * 16 + j;
            float v  = xc0[(size_t)kk * NPIX];
            __bf16 hv = (__bf16)v;
            a_hi[j] = hv;
            a_lo[j] = (__bf16)(v - (float)hv);
        }

        const __bf16* bhc = bh + ((size_t)(cc * 8) * 32 + lane) * 16;
        const __bf16* blc = bl + ((size_t)(cc * 8) * 32 + lane) * 16;

        #pragma unroll
        for (int t = 0; t < 8; ++t) {                // cluster tiles
            v16bf b_hi = *(const v16bf*)(bhc + (size_t)t * 32 * 16);
            v16bf b_lo = *(const v16bf*)(blc + (size_t)t * 32 * 16);
            // hi*hi + hi*lo + lo*hi  (drop lo*lo) ~ fp32-accurate dot
            acc[t] = __builtin_amdgcn_wmma_f32_16x16x32_bf16(
                false, a_hi, false, b_hi, (short)0, acc[t], false, false);
            acc[t] = __builtin_amdgcn_wmma_f32_16x16x32_bf16(
                false, a_hi, false, b_lo, (short)0, acc[t], false, false);
            acc[t] = __builtin_amdgcn_wmma_f32_16x16x32_bf16(
                false, a_lo, false, b_hi, (short)0, acc[t], false, false);
        }
    }

    // ---- argmin over 128 clusters of score = c2[k] - 2*xc ----
    // D layout: lane col = cluster%16; VGPR j = pixel (g*8 + j).
    float minv[8];
    int   mini[8];
    #pragma unroll
    for (int j = 0; j < 8; ++j) { minv[j] = 3.4e38f; mini[j] = 0; }

    #pragma unroll
    for (int t = 0; t < 8; ++t) {
        const int k = t * 16 + col;
        const float ck = c2[k];
        #pragma unroll
        for (int j = 0; j < 8; ++j) {
            float s = ck - 2.0f * acc[t][j];
            if (s < minv[j]) { minv[j] = s; mini[j] = k; }   // t asc -> first-min
        }
    }

    const int blk_base = (h >> 3) * 8 + (w0 >> 3);
    const float invc = 1.0f / 64.0f;

    #pragma unroll
    for (int j = 0; j < 8; ++j) {
        // min-reduce across the 16 lanes of this half (masks stay in-group)
        #pragma unroll
        for (int mask = 1; mask < 16; mask <<= 1) {
            float ov = __shfl_xor(minv[j], mask, 32);
            int   oi = __shfl_xor(mini[j], mask, 32);
            if (ov < minv[j] || (ov == minv[j] && oi < mini[j])) {
                minv[j] = ov; mini[j] = oi;
            }
        }
        // one writer per pixel: lane 'j' of each half-group
        if (col == j) {
            const int blk  = blk_base + g;           // pixel m = g*8 + j
            const int bin  = mini[j] + 1;            // code+1 in [1,128]
            atomicAdd(out + ((size_t)n * 64 + blk) * 129 + bin, invc);
        }
    }
}

// ---------------------------------------------------------------------------
extern "C" void kernel_launch(void* const* d_in, const int* in_sizes, int n_in,
                              void* d_out, int out_size, void* d_ws, size_t ws_size,
                              hipStream_t stream) {
    const float* x       = (const float*)d_in[0];   // (32,256,64,64) fp32
    const float* centers = (const float*)d_in[1];   // (128,256) fp32
    float* out = (float*)d_out;                     // (32,64,129) fp32

    float*  c2 = (float*)d_ws;
    __bf16* bhp = (__bf16*)((char*)d_ws + WS_BH_OFF);
    __bf16* blp = (__bf16*)((char*)d_ws + WS_BL_OFF);

    grb_prep_c2<<<1, 128, 0, stream>>>(centers, c2);
    grb_prep_bpack<<<128, 256, 0, stream>>>(centers, bhp, blp);
    grb_zero_out<<<(OUTEL + 255) / 256, 256, 0, stream>>>(out, OUTEL);
    // 8192 pixel tiles, 8 waves (256 threads) per block -> 1024 blocks
    grb_vq_main<<<1024, 256, 0, stream>>>(x, c2, bhp, blp, out);
}
